// DecisionTransformer_76527727280433
// MI455X (gfx1250) — matrix operand order repaired and checked by reading, not compiled, not run on hardware
//
#include <hip/hip_runtime.h>
#include <cstddef>

// ---------------- model dims ----------------
#define Bsz   64
#define Lsz   30
#define Dm    512
#define Hh    8
#define NBk   6
#define ASz   18
#define Tt    (3*Lsz)        // 90
#define HDim  (Dm/Hh)        // 64
#define ROWS  (Bsz*Tt)       // 5760
#define BLr   (Bsz*Lsz)      // 1920
#define CONVF 3136

typedef float  v8f   __attribute__((ext_vector_type(8)));
typedef __bf16 v16bf __attribute__((ext_vector_type(16)));

union Frag { v16bf v; unsigned u[8]; };

__device__ __forceinline__ unsigned short f2bf(float f) {
  unsigned u = __float_as_uint(f);
  return (unsigned short)((u + 0x7FFFu + ((u >> 16) & 1u)) >> 16);
}

// async global->LDS (gfx1250) if the toolchain exposes it; guarded fallback otherwise
#if defined(__has_builtin)
#if __has_builtin(__builtin_amdgcn_global_load_async_to_lds_b128) && \
    __has_builtin(__builtin_amdgcn_s_wait_asynccnt)
#define HAVE_ASYNC_LDS 1
#endif
#endif

#ifdef HAVE_ASYNC_LDS
typedef int b128v __attribute__((vector_size(16)));                 // GCC-style vec, per builtin sig
typedef __attribute__((address_space(1))) b128v b128v_g;            // global
typedef __attribute__((address_space(3))) b128v b128v_l;            // LDS
#endif

// ------------- weight convert+transpose: W(KxN f32) -> Wt(NxK bf16) ----------
__global__ __launch_bounds__(256)
void wcvt_kernel(const float* __restrict__ src, unsigned short* __restrict__ dst,
                 int K, int N)
{
  int idx = blockIdx.x * 256 + threadIdx.x;
  if (idx >= K * N) return;
  int n = idx / K, k = idx - n * K;
  dst[idx] = f2bf(src[(size_t)k * N + n]);
}

// ---------------- WMMA GEMM: C = act(A(MxK) @ B(KxN) + bias) + resid ---------
// A: f32 row-major. Bt: bf16, PRE-TRANSPOSED to [N][K]. act: 0 none,1 tanh,2 gelu
#define BM 128
#define BN 256
#define BK 32

__global__ __launch_bounds__(256)
void gemm_bf16_kernel(const float* __restrict__ A, const unsigned short* __restrict__ Bt,
                      const float* __restrict__ bias, const float* __restrict__ resid,
                      float* __restrict__ C, int M, int N, int K, int act)
{
  __shared__ __align__(16) unsigned short As[BM][BK];   // bf16 bits, [m][k]
  __shared__ __align__(16) unsigned short Bs[BN][BK];   // bf16 bits, [n][k]

  const int tid  = threadIdx.x;
  const int wid  = tid >> 5;
  const int lane = tid & 31;
  const int hl   = lane >> 4;
  const int lm   = lane & 15;
  const int m0blk = blockIdx.y * BM;
  const int n0blk = blockIdx.x * BN;
  const int wm = (wid & 1) * 64;   // 2 waves across M, wave tile 64 rows
  const int wn = (wid >> 1) * 64;  // 4 waves across N, wave tile 64 cols
  const bool fullN = (n0blk + BN) <= N;

  v8f acc[4][4] = {};

  for (int k0 = 0; k0 < K; k0 += BK) {
    // stage A tile: pack 2 k-adjacent f32 -> one b32 of bf16 pair
    for (int i = tid; i < BM * BK / 2; i += 256) {
      int m = i >> 4;
      int kk = (i & 15) << 1;
      int gm = m0blk + m;
      unsigned lo = 0, hi = 0;
      if (gm < M) {
        const float* ap = &A[(size_t)gm * K + k0 + kk];
        lo = f2bf(ap[0]); hi = f2bf(ap[1]);
      }
      *(unsigned*)&As[m][kk] = lo | (hi << 16);
    }
    // stage B tile: 16B chunks of pre-transposed bf16 weights
#ifdef HAVE_ASYNC_LDS
    if (fullN) {
      // async DMA straight into LDS, no VGPR round-trip (ASYNCcnt-tracked)
      for (int i = tid; i < BN * 4; i += 256) {
        int n = i >> 2, c = (i & 3) << 3;
        const unsigned short* gp = &Bt[(size_t)(n0blk + n) * K + k0 + c];
        __builtin_amdgcn_global_load_async_to_lds_b128(
            (b128v_g*)(void*)gp, (b128v_l*)(void*)&Bs[n][c], 0, 0);
      }
      __builtin_amdgcn_s_wait_asynccnt(0);
    } else
#endif
    {
      for (int i = tid; i < BN * 4; i += 256) {
        int n = i >> 2, c = (i & 3) << 3;            // 4 chunks of 8 bf16 per row
        int gn = n0blk + n;
        uint4 val = make_uint4(0u, 0u, 0u, 0u);
        if (gn < N) val = *(const uint4*)&Bt[(size_t)gn * K + k0 + c];
        *(uint4*)&Bs[n][c] = val;
      }
    }
    if (k0 + BK < K) {  // speculative prefetch of next A tile -> global_prefetch_b8
      __builtin_prefetch(&A[(size_t)(m0blk + (tid & 127)) * K + (k0 + BK)], 0, 1);
    }
    __syncthreads();

    Frag af[4], bfr[4];
    #pragma unroll
    for (int i = 0; i < 4; ++i) {
      int mrow = wm + 16 * i + lm;
      #pragma unroll
      for (int p = 0; p < 8; ++p) {
        int kp = 2 * (p & 3) + 8 * hl + 16 * (p >> 2);   // ISA 16-bit A 16x32 layout
        af[i].u[p] = *(const unsigned*)&As[mrow][kp];
      }
    }
    #pragma unroll
    for (int j = 0; j < 4; ++j) {
      int ncol = wn + 16 * j + lm;
      #pragma unroll
      for (int p = 0; p < 8; ++p) {
        int kb = 2 * (p & 3) + 8 * hl + 16 * (p >> 2);   // ISA 16-bit B 32x16 layout
        bfr[j].u[p] = *(const unsigned*)&Bs[ncol][kb];
      }
    }
    #pragma unroll
    for (int i = 0; i < 4; ++i)
      #pragma unroll
      for (int j = 0; j < 4; ++j)
        acc[i][j] = __builtin_amdgcn_wmma_f32_16x16x32_bf16(
            false, af[i].v, false, bfr[j].v, (short)0, acc[i][j], false, false);
    __syncthreads();
  }

  // epilogue: C layout — VGPR v holds M = v + 8*half, N = lane%16
  #pragma unroll
  for (int i = 0; i < 4; ++i)
    #pragma unroll
    for (int j = 0; j < 4; ++j)
      #pragma unroll
      for (int v = 0; v < 8; ++v) {
        int row = m0blk + wm + 16 * i + v + 8 * hl;
        int col = n0blk + wn + 16 * j + lm;
        if (row < M && col < N) {
          float val = acc[i][j][v];
          if (bias) val += bias[col];
          if (act == 1) val = tanhf(val);
          else if (act == 2) val = 0.5f * val * (1.f + erff(val * 0.70710678118f));
          if (resid) val += resid[(size_t)row * N + col];
          C[(size_t)row * N + col] = val;
        }
      }
}

// ---------------- conv stem (fp32 direct, fused relu) ------------------------
__global__ __launch_bounds__(256)
void conv1_kernel(const float* __restrict__ in, const float* __restrict__ w,
                  const float* __restrict__ bias, float* __restrict__ out)
{
  int idx = blockIdx.x * 256 + threadIdx.x;
  if (idx >= BLr * 20 * 20 * 32) return;
  int co = idx & 31; int t = idx >> 5;
  int ox = t % 20; t /= 20; int oy = t % 20; int img = t / 20;
  const float* ip = in + (size_t)img * 84 * 84 * 4;
  float acc = bias[co];
  for (int kh = 0; kh < 8; ++kh)
    for (int kw = 0; kw < 8; ++kw) {
      const float* r  = ip + (size_t)((oy * 4 + kh) * 84 + (ox * 4 + kw)) * 4;
      const float* wr = w + (size_t)((kh * 8 + kw) * 4) * 32 + co;
      #pragma unroll
      for (int ci = 0; ci < 4; ++ci)
        acc += r[ci] * 0.003921568627451f * wr[ci * 32];
    }
  out[idx] = fmaxf(acc, 0.f);
}

__global__ __launch_bounds__(256)
void conv2_kernel(const float* __restrict__ in, const float* __restrict__ w,
                  const float* __restrict__ bias, float* __restrict__ out)
{
  int idx = blockIdx.x * 256 + threadIdx.x;
  if (idx >= BLr * 9 * 9 * 64) return;
  int co = idx & 63; int t = idx >> 6;
  int ox = t % 9; t /= 9; int oy = t % 9; int img = t / 9;
  const float* ip = in + (size_t)img * 20 * 20 * 32;
  float acc = bias[co];
  for (int kh = 0; kh < 4; ++kh)
    for (int kw = 0; kw < 4; ++kw) {
      const float* r  = ip + (size_t)((oy * 2 + kh) * 20 + (ox * 2 + kw)) * 32;
      const float* wr = w + (size_t)((kh * 4 + kw) * 32) * 64 + co;
      for (int ci = 0; ci < 32; ++ci)
        acc += r[ci] * wr[ci * 64];
    }
  out[idx] = fmaxf(acc, 0.f);
}

__global__ __launch_bounds__(256)
void conv3_kernel(const float* __restrict__ in, const float* __restrict__ w,
                  const float* __restrict__ bias, float* __restrict__ out)
{
  int idx = blockIdx.x * 256 + threadIdx.x;
  if (idx >= BLr * 7 * 7 * 64) return;
  int co = idx & 63; int t = idx >> 6;
  int ox = t % 7; t /= 7; int oy = t % 7; int img = t / 7;
  const float* ip = in + (size_t)img * 9 * 9 * 64;
  float acc = bias[co];
  for (int kh = 0; kh < 3; ++kh)
    for (int kw = 0; kw < 3; ++kw) {
      const float* r  = ip + (size_t)((oy + kh) * 9 + (ox + kw)) * 64;
      const float* wr = w + (size_t)((kh * 3 + kw) * 64) * 64 + co;
      for (int ci = 0; ci < 64; ++ci)
        acc += r[ci] * wr[ci * 64];
    }
  out[idx] = fmaxf(acc, 0.f);
}

// ---------------- token assembly ---------------------------------------------
__global__ __launch_bounds__(256)
void tokens_kernel(const float* __restrict__ rtgs, const int* __restrict__ actions,
                   const int* __restrict__ timesteps, const float* __restrict__ W_rtg,
                   const float* __restrict__ b_rtg, const float* __restrict__ emb_a,
                   const float* __restrict__ abs_pos, const float* __restrict__ rel_pos,
                   const float* __restrict__ st_e, float* __restrict__ x)
{
  int idx = blockIdx.x * 256 + threadIdx.x;
  if (idx >= Bsz * Lsz * Dm) return;
  int d = idx & (Dm - 1); int t = idx >> 9;
  int l = t % Lsz; int b = t / Lsz;
  int ts = timesteps[b];
  float absd = abs_pos[(size_t)ts * Dm + d];
  float rtg = tanhf(rtgs[b * Lsz + l] * W_rtg[d] + b_rtg[d]);
  int a = actions[b * Lsz + l];
  float ae = tanhf(emb_a[(size_t)a * Dm + d]);
  float se = st_e[(size_t)(b * Lsz + l) * Dm + d];   // already tanh'd in GEMM epilogue
  int t0 = 3 * l;
  size_t base = (size_t)(b * Tt + t0) * Dm + d;
  x[base]                  = rtg + rel_pos[(size_t)(t0 + 0) * Dm + d] + absd;
  x[base + Dm]             = se  + rel_pos[(size_t)(t0 + 1) * Dm + d] + absd;
  x[base + 2 * (size_t)Dm] = ae  + rel_pos[(size_t)(t0 + 2) * Dm + d] + absd;
}

// ---------------- layernorm (one block per row, D=512) -----------------------
__global__ __launch_bounds__(256)
void layernorm_kernel(const float* __restrict__ x, const float* __restrict__ g,
                      const float* __restrict__ b, float* __restrict__ out)
{
  int row = blockIdx.x;
  const float* xp = x + (size_t)row * Dm;
  int t = threadIdx.x;
  float v0 = xp[t], v1 = xp[t + 256];
  __shared__ float red[8], red2[8];
  float s = v0 + v1;
  for (int o = 16; o > 0; o >>= 1) s += __shfl_xor(s, o);
  if ((t & 31) == 0) red[t >> 5] = s;
  __syncthreads();
  if (t < 8) {
    float z = red[t];
    for (int o = 4; o > 0; o >>= 1) z += __shfl_xor(z, o);
    if (t == 0) red[0] = z;
  }
  __syncthreads();
  float mean = red[0] * (1.f / Dm);
  float d0 = v0 - mean, d1 = v1 - mean;
  float q = d0 * d0 + d1 * d1;
  for (int o = 16; o > 0; o >>= 1) q += __shfl_xor(q, o);
  if ((t & 31) == 0) red2[t >> 5] = q;
  __syncthreads();
  if (t < 8) {
    float z = red2[t];
    for (int o = 4; o > 0; o >>= 1) z += __shfl_xor(z, o);
    if (t == 0) red2[0] = z;
  }
  __syncthreads();
  float rstd = rsqrtf(red2[0] * (1.f / Dm) + 1e-3f);
  out[(size_t)row * Dm + t]       = d0 * rstd * g[t] + b[t];
  out[(size_t)row * Dm + t + 256] = d1 * rstd * g[t + 256] + b[t + 256];
}

// ---------------- attention: s = K@Q^T (faithful), causal softmax, o = S@V ---
__global__ __launch_bounds__(256)
void attn_kernel(const float* __restrict__ kk, const float* __restrict__ qq,
                 const float* __restrict__ vv, float* __restrict__ x)
{
  int bh = blockIdx.x; int b = bh >> 3; int h = bh & 7;
  __shared__ float S[Tt][Tt + 1];
  int tid = threadIdx.x;
  size_t hoff = (size_t)h * HDim;
  const float scale = 0.125f;  // 1/sqrt(64)

  for (int p = tid; p < Tt * Tt; p += 256) {
    int i = p / Tt, j = p % Tt;
    float a = 0.f;
    if (j <= i) {
      const float* kp = kk + ((size_t)(b * Tt + i)) * Dm + hoff;
      const float* qp = qq + ((size_t)(b * Tt + j)) * Dm + hoff;
      for (int d = 0; d < HDim; ++d) a += kp[d] * qp[d];
      a *= scale;
    }
    S[i][j] = a;
  }
  __syncthreads();
  if (tid < Tt) {
    int i = tid;
    float mx = -1e30f;
    for (int j = 0; j <= i; ++j) mx = fmaxf(mx, S[i][j]);
    float sum = 0.f;
    for (int j = 0; j <= i; ++j) { float e = expf(S[i][j] - mx); S[i][j] = e; sum += e; }
    float inv = 1.f / sum;
    for (int j = 0; j <= i; ++j) S[i][j] *= inv;
    for (int j = i + 1; j < Tt; ++j) S[i][j] = 0.f;
  }
  __syncthreads();
  for (int p = tid; p < Tt * HDim; p += 256) {
    int i = p >> 6, d = p & 63;
    float a = 0.f;
    for (int j = 0; j <= i; ++j)
      a += S[i][j] * vv[((size_t)(b * Tt + j)) * Dm + hoff + d];
    size_t off = ((size_t)(b * Tt + i)) * Dm + hoff + d;
    x[off] += a;   // residual fused
  }
}

// ---------------- gather state rows (t = 3l+1) -------------------------------
__global__ __launch_bounds__(256)
void gather_kernel(const float* __restrict__ hbuf, float* __restrict__ hs)
{
  int idx = blockIdx.x * 256 + threadIdx.x;
  if (idx >= BLr * Dm) return;
  int d = idx & (Dm - 1); int r = idx >> 9;
  int b = r / Lsz, l = r % Lsz;
  hs[idx] = hbuf[((size_t)(b * Tt + 3 * l + 1)) * Dm + d];
}

// ---------------- host side --------------------------------------------------
extern "C" void kernel_launch(void* const* d_in, const int* in_sizes, int n_in,
                              void* d_out, int out_size, void* d_ws, size_t ws_size,
                              hipStream_t stream) {
  const float* rtgs      = (const float*)d_in[0];
  const float* states    = (const float*)d_in[1];
  const int*   actions   = (const int*)d_in[2];
  const int*   timesteps = (const int*)d_in[3];
  const float* W_rtg     = (const float*)d_in[4];
  const float* b_rtg     = (const float*)d_in[5];
  const float* conv1_w   = (const float*)d_in[6];
  const float* conv1_b   = (const float*)d_in[7];
  const float* conv2_w   = (const float*)d_in[8];
  const float* conv2_b   = (const float*)d_in[9];
  const float* conv3_w   = (const float*)d_in[10];
  const float* conv3_b   = (const float*)d_in[11];
  const float* dense_s_w = (const float*)d_in[12];
  const float* dense_s_b = (const float*)d_in[13];
  const float* emb_a     = (const float*)d_in[14];
  const float* abs_pos   = (const float*)d_in[15];
  const float* rel_pos   = (const float*)d_in[16];
  const float* ln1_g     = (const float*)d_in[17];
  const float* ln1_b     = (const float*)d_in[18];
  const float* Wk        = (const float*)d_in[19];
  const float* bk        = (const float*)d_in[20];
  const float* Wq        = (const float*)d_in[21];
  const float* bq        = (const float*)d_in[22];
  const float* Wv        = (const float*)d_in[23];
  const float* bv        = (const float*)d_in[24];
  const float* ln2_g     = (const float*)d_in[25];
  const float* ln2_b     = (const float*)d_in[26];
  const float* W1        = (const float*)d_in[27];
  const float* b1        = (const float*)d_in[28];
  const float* W2        = (const float*)d_in[29];
  const float* b2        = (const float*)d_in[30];
  const float* lnf_g     = (const float*)d_in[31];
  const float* lnf_b     = (const float*)d_in[32];
  const float* W_head    = (const float*)d_in[33];
  (void)in_sizes; (void)n_in; (void)out_size; (void)ws_size;

  // ---- workspace layout (float units); regions reused along the timeline ----
  // phase 1: conv1o[0,24.576M) conv2o[24.576M,34.529M) conv3o[34.529M,40.55M)
  // phase 2 (after conv2): bf16 weights reuse conv1o region; acts follow.
  float* ws     = (float*)d_ws;
  float* conv1o = ws;                 // 24,576,000
  float* conv2o = ws + 24576000;      //  9,953,280
  float* conv3o = ws + 34529280;      //  6,021,120 (live until dense_s GEMM)

  unsigned short* wbf = (unsigned short*)d_ws;     // bf16 region (reuses conv1o)
  unsigned short* dsw_t = wbf;                     // 512 x 3136       = 1,605,632
  unsigned short* wk_t  = wbf + 1605632;           // 6 x 512 x 512    = 1,572,864
  unsigned short* wq_t  = wbf + 3178496;
  unsigned short* wv_t  = wbf + 4751360;
  unsigned short* w1_t  = wbf + 6324224;           // 6 x 2048 x 512   = 6,291,456
  unsigned short* w2_t  = wbf + 12615680;          // 6 x 512 x 2048   = 6,291,456
  unsigned short* wh_t  = wbf + 18907136;          // 18 x 512         = 9,216
                                                   // ends 18,916,352 ushort = 9,458,176 f
  float* st  = ws + 9458176;          //    983,040
  float* xb  = ws + 10441216;         //  2,949,120
  float* hb  = ws + 13390336;         //  2,949,120
  float* kb  = ws + 16339456;         //  2,949,120
  float* qb  = ws + 19288576;         //  2,949,120
  float* vb  = ws + 22237696;         //  2,949,120
  float* ffb = ws + 25186816;         // 11,796,480 (overlaps dead conv3o region)
  float* hsb = ws + 36983296;         //    983,040
  float* outp = (float*)d_out;

  auto gemm = [&](const float* A, const unsigned short* Bt, const float* bias,
                  const float* resid, float* C, int M, int N, int K, int act) {
    dim3 g((N + BN - 1) / BN, (M + BM - 1) / BM);
    gemm_bf16_kernel<<<g, 256, 0, stream>>>(A, Bt, bias, resid, C, M, N, K, act);
  };
  auto wcvt = [&](const float* src, unsigned short* dst, int K, int N) {
    wcvt_kernel<<<(K * N + 255) / 256, 256, 0, stream>>>(src, dst, K, N);
  };

  // conv stem
  conv1_kernel<<<(BLr * 20 * 20 * 32 + 255) / 256, 256, 0, stream>>>(states, conv1_w, conv1_b, conv1o);
  conv2_kernel<<<(BLr * 9 * 9 * 64 + 255) / 256, 256, 0, stream>>>(conv1o, conv2_w, conv2_b, conv2o);
  conv3_kernel<<<(BLr * 7 * 7 * 64 + 255) / 256, 256, 0, stream>>>(conv2o, conv3_w, conv3_b, conv3o);

  // weight convert+transpose to bf16 (conv1o/conv2o regions are dead now)
  wcvt(dense_s_w, dsw_t, CONVF, Dm);
  for (int i = 0; i < NBk; ++i) {
    wcvt(Wk + (size_t)i * Dm * Dm, wk_t + (size_t)i * Dm * Dm, Dm, Dm);
    wcvt(Wq + (size_t)i * Dm * Dm, wq_t + (size_t)i * Dm * Dm, Dm, Dm);
    wcvt(Wv + (size_t)i * Dm * Dm, wv_t + (size_t)i * Dm * Dm, Dm, Dm);
    wcvt(W1 + (size_t)i * Dm * 4 * Dm, w1_t + (size_t)i * Dm * 4 * Dm, Dm, 4 * Dm);
    wcvt(W2 + (size_t)i * Dm * 4 * Dm, w2_t + (size_t)i * Dm * 4 * Dm, 4 * Dm, Dm);
  }
  wcvt(W_head, wh_t, Dm, ASz);

  // state embedding: tanh(conv_flat @ dense_s_w + b)   (WMMA)
  gemm(conv3o, dsw_t, dense_s_b, nullptr, st, BLr, Dm, CONVF, 1);
  // assemble tokens + positions
  tokens_kernel<<<(Bsz * Lsz * Dm + 255) / 256, 256, 0, stream>>>(
      rtgs, actions, timesteps, W_rtg, b_rtg, emb_a, abs_pos, rel_pos, st, xb);

  for (int i = 0; i < NBk; ++i) {
    const float* g1 = ln1_g + i * Dm; const float* be1 = ln1_b + i * Dm;
    const float* g2 = ln2_g + i * Dm; const float* be2 = ln2_b + i * Dm;
    size_t wo = (size_t)i * Dm * Dm;
    size_t wo4 = (size_t)i * Dm * 4 * Dm;
    layernorm_kernel<<<ROWS, 256, 0, stream>>>(xb, g1, be1, hb);
    gemm(hb, wk_t + wo, bk + i * Dm, nullptr, kb, ROWS, Dm, Dm, 0);
    gemm(hb, wq_t + wo, bq + i * Dm, nullptr, qb, ROWS, Dm, Dm, 0);
    gemm(hb, wv_t + wo, bv + i * Dm, nullptr, vb, ROWS, Dm, Dm, 0);
    attn_kernel<<<Bsz * Hh, 256, 0, stream>>>(kb, qb, vb, xb);   // x += attn
    layernorm_kernel<<<ROWS, 256, 0, stream>>>(xb, g2, be2, hb);
    gemm(hb, w1_t + wo4, b1 + i * 4 * Dm, nullptr, ffb, ROWS, 4 * Dm, Dm, 2);
    gemm(ffb, w2_t + wo4, b2 + i * Dm, xb, xb, ROWS, Dm, 4 * Dm, 0);
  }

  layernorm_kernel<<<ROWS, 256, 0, stream>>>(xb, lnf_g, lnf_b, hb);
  gather_kernel<<<(BLr * Dm + 255) / 256, 256, 0, stream>>>(hb, hsb);
  gemm(hsb, wh_t, nullptr, nullptr, outp, BLr, ASz, Dm, 0);
}
